// RelativeGlobalAttention_87677462381235
// MI455X (gfx1250) — matrix operand order repaired and checked by reading, not compiled, not run on hardware
//
#include <hip/hip_runtime.h>

// ---------------------------------------------------------------------------
// Relative Global Attention, CDNA5 (gfx1250), wave32, bf16 WMMA w/ f32 accum,
// async global->LDS staging, double-buffered mainloops, NT hints on the
// 268 MB attn stream (L2 = 192 MB). B=2, H=8, L=M=2048, D=512, dh=64.
// ---------------------------------------------------------------------------

typedef __attribute__((ext_vector_type(16))) __bf16 v16bf;
typedef __attribute__((ext_vector_type(8)))  float  v8f;

#define HEADS 8
#define SEQL  2048
#define DMODEL 512
#define DHEAD 64
#define NBATCH 2
#define MREL SEQL
#define NBH (NBATCH * HEADS)

// ---------------- WMMA fragment helpers (contiguous k = 16*(lane/16) + e in A and B) -----------

__device__ inline v16bf load_a16(const __bf16* s, int stride) {
  const int lane = threadIdx.x & 31;
  const int row = lane & 15;
  const int khalf = lane & 16;  // 0 or 16
  v16bf a;
#pragma unroll
  for (int e = 0; e < 16; ++e) {
    const int k = khalf + e;
    a[e] = s[row * stride + k];
  }
  return a;
}

// B 32x16, source row-major B[k][n] at s[k*stride+n].
__device__ inline v16bf load_b16_row(const __bf16* s, int stride) {
  const int lane = threadIdx.x & 31;
  const int n = lane & 15;
  const int kbase = lane & 16;
  v16bf b;
#pragma unroll
  for (int e = 0; e < 16; ++e) b[e] = s[(kbase + e) * stride + n];
  return b;
}

// B 32x16, source "n-major": B[k][n] at s[n*stride+k] (vectorizes to b128).
__device__ inline v16bf load_b16_col(const __bf16* s, int stride) {
  const int lane = threadIdx.x & 31;
  const int n = lane & 15;
  const int kbase = lane & 16;
  v16bf b;
#pragma unroll
  for (int e = 0; e < 16; ++e) b[e] = s[n * stride + kbase + e];
  return b;
}

__device__ inline v8f wmma_bf16(v16bf a, v16bf b, v8f c) {
  return __builtin_amdgcn_wmma_f32_16x16x32_bf16(
      false, a, false, b, (short)0, c, false, false);
}

// ---------------- gfx1250 async global->LDS copy (CDNA5 ISA) ------

__device__ inline unsigned lds_off_of(const void* p) {
  return (unsigned)(unsigned long long)p;  // low 32 bits = LDS byte offset
}

__device__ inline void async_ld_b128(const void* gaddr, unsigned lds_byte_off) {
  asm volatile("global_load_async_to_lds_b128 %0, %1, off"
               :: "v"(lds_byte_off), "v"(gaddr) : "memory");
}

__device__ inline void async_wait0() {
  asm volatile("s_wait_asynccnt 0" ::: "memory");
}

// ---------------- f32 -> bf16 convert (for E) ------------------------------

__global__ __launch_bounds__(256) void k_f32_to_bf16(const float* __restrict__ src,
                                                     __bf16* __restrict__ dst, int n) {
  const int i = blockIdx.x * 256 + threadIdx.x;
  if (i < n) dst[i] = (__bf16)src[i];
}

// ---------------- generic 4096x512x512 GEMM (+bias), double buffered -------

#define PROJ_ASTR 34
#define PROJ_BSTR 66

template <bool HEAD_LAYOUT>
__global__ __launch_bounds__(256) void k_gemm512(const float* __restrict__ X,
                                                 const float* __restrict__ W,
                                                 const float* __restrict__ bias,
                                                 void* __restrict__ dstv) {
  __shared__ __bf16 As[2][128 * PROJ_ASTR];
  __shared__ __bf16 Bs[2][32 * PROJ_BSTR];
  const int t = threadIdx.x;
  const int wave = t >> 5, lane = t & 31;
  const int n0 = blockIdx.x * 128;
  const int j0 = blockIdx.y * 64;
  const int arow = t >> 1, acb = (t & 1) * 16;
  const int brow = t >> 3, bcb = (t & 7) * 8;

  auto stage = [&](int kt, int buf) {
    const float* src = X + (size_t)(n0 + arow) * DMODEL + kt + acb;
    __builtin_prefetch(src + 32, 0, 0);  // -> global_prefetch_b8
#pragma unroll
    for (int e = 0; e < 16; ++e) As[buf][arow * PROJ_ASTR + acb + e] = (__bf16)src[e];
    const float* wsrc = W + (size_t)(kt + brow) * DMODEL + j0 + bcb;
    __builtin_prefetch(wsrc + 32 * DMODEL, 0, 0);
#pragma unroll
    for (int e = 0; e < 8; ++e) Bs[buf][brow * PROJ_BSTR + bcb + e] = (__bf16)wsrc[e];
  };

  v8f acc[4] = {};
  stage(0, 0);
  __syncthreads();

  for (int kt = 0; kt < DMODEL; kt += 32) {
    const int cur = (kt >> 5) & 1;
    if (kt + 32 < DMODEL) stage(kt + 32, cur ^ 1);
    const v16bf a = load_a16(&As[cur][(wave * 16) * PROJ_ASTR], PROJ_ASTR);
#pragma unroll
    for (int nt = 0; nt < 4; ++nt) {
      const v16bf b = load_b16_row(&Bs[cur][nt * 16], PROJ_BSTR);
      acc[nt] = wmma_bf16(a, b, acc[nt]);
    }
    __syncthreads();  // publishes next buffer, retires reads of current
  }

  const int rhi = (lane & 16) >> 1;
  const int col = lane & 15;
#pragma unroll
  for (int nt = 0; nt < 4; ++nt) {
    const int j = j0 + nt * 16 + col;
    const float bv = bias[j];
#pragma unroll
    for (int r = 0; r < 8; ++r) {
      const int n = n0 + wave * 16 + r + rhi;
      const float vv = acc[nt][r] + bv;
      if constexpr (HEAD_LAYOUT) {
        const int bb = n >> 11, l = n & (SEQL - 1);
        const int h = j >> 6, dd = j & 63;
        ((__bf16*)dstv)[(((size_t)(bb * HEADS + h)) * SEQL + l) * DHEAD + dd] = (__bf16)vv;
      } else {
        ((float*)dstv)[(size_t)n * DMODEL + j] = vv;
      }
    }
  }
}

// ---------------- fused logits: (QK^T + Srel(QE skew)) / sqrt(dh) ----------
// Srel[i,j] = QE[i, m], m = j + (M-1) - i for j<=i else 0. For a 16x128 block
// the m's span a 143-wide band from mb0 = j0 + M - 16 - i0: compute that QE
// band with WMMA into LDS, gather per element, single NT write of attn.

#define QSTR 72
#define KSTR 72
#define ESTR 72
#define QESTR 161
#define EBAND 160

__global__ __launch_bounds__(256) void k_logits_fused(const __bf16* __restrict__ Qbf,
                                                      const __bf16* __restrict__ Kbf,
                                                      const __bf16* __restrict__ Ebf,
                                                      float* __restrict__ attn) {
  __shared__ __bf16 Qs[16 * QSTR];
  __shared__ __bf16 Ks[128 * KSTR];
  __shared__ __bf16 Es[EBAND * ESTR];
  __shared__ float  QEs[16 * QESTR];

  const int t = threadIdx.x, wave = t >> 5, lane = t & 31;
  const int i0 = blockIdx.x * 16;
  const int j0g = blockIdx.y * 128;
  const int bh = blockIdx.z;
  const int mb0 = j0g + MREL - 16 - i0;  // 16-aligned, >= 0

  {  // Q tile 16x64
    const int row = t >> 4, cb = (t & 15) * 4;
    const __bf16* src = Qbf + ((size_t)bh * SEQL + i0 + row) * DHEAD + cb;
#pragma unroll
    for (int e = 0; e < 4; ++e) Qs[row * QSTR + cb + e] = src[e];
  }
  {  // K tile 128x64 via async global->LDS (4 x b128 per thread)
    const int jrow = t >> 1, cb2 = (t & 1) * 32;
    const __bf16* ksrc = Kbf + ((size_t)bh * SEQL + j0g + jrow) * DHEAD + cb2;
    const unsigned koff = lds_off_of(&Ks[jrow * KSTR + cb2]);
#pragma unroll
    for (int c = 0; c < 4; ++c) async_ld_b128(ksrc + c * 8, koff + c * 16);
  }
  // E band 160x64; rows with m >= SEQL are zero (feed only masked lanes).
  for (int r0 = 0; r0 < EBAND; r0 += 128) {
    const int er = r0 + (t >> 1);
    if (er < EBAND) {
      const int cb2 = (t & 1) * 32;
      const int m = mb0 + er;
      if (m < SEQL) {
        const __bf16* esrc = Ebf + (size_t)m * DHEAD + cb2;
        const unsigned eoff = lds_off_of(&Es[er * ESTR + cb2]);
#pragma unroll
        for (int c = 0; c < 4; ++c) async_ld_b128(esrc + c * 8, eoff + c * 16);
      } else {
#pragma unroll
        for (int e = 0; e < 32; ++e) Es[er * ESTR + cb2 + e] = (__bf16)0.0f;
      }
    }
  }
  async_wait0();
  __syncthreads();

  // QE band: 10 tiles of 16 cols across 8 waves (wave-uniform trip counts).
  for (int tt = wave; tt < EBAND / 16; tt += 8) {
    v8f qe = {};
#pragma unroll
    for (int kt = 0; kt < DHEAD; kt += 32) {
      const v16bf a = load_a16(&Qs[kt], QSTR);
      const v16bf b = load_b16_col(&Es[(tt * 16) * ESTR + kt], ESTR);
      qe = wmma_bf16(a, b, qe);
    }
    const int rhi = (lane & 16) >> 1;
    const int col = lane & 15;
#pragma unroll
    for (int r = 0; r < 8; ++r)
      QEs[(r + rhi) * QESTR + tt * 16 + col] = qe[r];
  }
  __syncthreads();

  // QK^T tile: wave handles j-subtile [j0g + wave*16, +16)
  v8f acc = {};
#pragma unroll
  for (int kt = 0; kt < DHEAD; kt += 32) {
    const v16bf a = load_a16(&Qs[kt], QSTR);
    const v16bf b = load_b16_col(&Ks[(wave * 16) * KSTR + kt], KSTR);
    acc = wmma_bf16(a, b, acc);
  }

  const int rhi = (lane & 16) >> 1;
  const int col = lane & 15;
#pragma unroll
  for (int r = 0; r < 8; ++r) {
    const int il = r + rhi;
    const int i = i0 + il;
    const int j = j0g + wave * 16 + col;
    const float srel = (j <= i) ? QEs[il * QESTR + (j + (MREL - 1) - i - mb0)] : 0.0f;
    __builtin_nontemporal_store((acc[r] + srel) * 0.125f,
                                &attn[((size_t)bh * SEQL + i) * MREL + j]);
  }
}

// ---------------- row softmax over 2048 columns, in place (NT stream) ------

__global__ __launch_bounds__(256) void k_softmax(float* __restrict__ attn) {
  float* row = attn + (size_t)blockIdx.x * MREL;
  const int t = threadIdx.x;
  float vals[8];
  float mx = -3.4e38f;
#pragma unroll
  for (int e = 0; e < 8; ++e) {
    vals[e] = __builtin_nontemporal_load(&row[t + 256 * e]);
    mx = fmaxf(mx, vals[e]);
  }
  __shared__ float red[256];
  red[t] = mx; __syncthreads();
  for (int s = 128; s > 0; s >>= 1) {
    if (t < s) red[t] = fmaxf(red[t], red[t + s]);
    __syncthreads();
  }
  mx = red[0]; __syncthreads();
  float sum = 0.0f;
#pragma unroll
  for (int e = 0; e < 8; ++e) { vals[e] = __expf(vals[e] - mx); sum += vals[e]; }
  red[t] = sum; __syncthreads();
  for (int s = 128; s > 0; s >>= 1) {
    if (t < s) red[t] += red[t + s];
    __syncthreads();
  }
  const float inv = 1.0f / red[0];
#pragma unroll
  for (int e = 0; e < 8; ++e)
    __builtin_nontemporal_store(vals[e] * inv, &row[t + 256 * e]);
}

// ---------------- attn @ V (K = 2048), double buffered ---------------------

#define AV_ASTR 40
#define AV_VSTR 72

__global__ __launch_bounds__(256) void k_av(const float* __restrict__ attn,
                                            const __bf16* __restrict__ Vbf,
                                            float* __restrict__ Obuf) {
  __shared__ __bf16 As[2][32 * AV_ASTR];
  __shared__ __bf16 Vs[2][32 * AV_VSTR];
  const int t = threadIdx.x, wave = t >> 5, lane = t & 31;
  const int i0 = blockIdx.x * 32;
  const int bh = blockIdx.y;
  const int wi = wave >> 2, wn = wave & 3;

  const int row = t >> 3;
  const int acb = (t & 7) * 4;   // attn stage: 4 f32 per thread
  const int vcb = (t & 7) * 8;   // V stage: 8 bf16 = 16 B per thread
  const unsigned voff[2] = { lds_off_of(&Vs[0][row * AV_VSTR + vcb]),
                             lds_off_of(&Vs[1][row * AV_VSTR + vcb]) };

  auto stage = [&](int jt, int buf) {
    const __bf16* vsrc = Vbf + ((size_t)bh * SEQL + jt + row) * DHEAD + vcb;
    async_ld_b128(vsrc, voff[buf]);
    const float* asrc = attn + ((size_t)bh * SEQL + i0 + row) * MREL + jt + acb;
#pragma unroll
    for (int e = 0; e < 4; ++e)
      As[buf][row * AV_ASTR + acb + e] = (__bf16)__builtin_nontemporal_load(asrc + e);
  };

  v8f acc = {};
  stage(0, 0);
  async_wait0();
  __syncthreads();

  for (int jt = 0; jt < MREL; jt += 32) {
    const int cur = (jt >> 5) & 1;
    if (jt + 32 < MREL) stage(jt + 32, cur ^ 1);  // async V overlaps the WMMA
    const v16bf a = load_a16(&As[cur][(wi * 16) * AV_ASTR], AV_ASTR);
    const v16bf b = load_b16_row(&Vs[cur][wn * 16], AV_VSTR);
    acc = wmma_bf16(a, b, acc);
    async_wait0();   // only the just-issued next-tile load is outstanding
    __syncthreads(); // publish next buffer / retire reads of current
  }

  const int rhi = (lane & 16) >> 1;
  const int col = lane & 15;
  const int bb = bh >> 3, h = bh & 7;
#pragma unroll
  for (int r = 0; r < 8; ++r) {
    const int i = i0 + wi * 16 + r + rhi;
    const int dd = wn * 16 + col;
    Obuf[((size_t)(bb * SEQL + i)) * DMODEL + h * DHEAD + dd] = acc[r];
  }
}

// ---------------- host-side orchestration ----------------------------------

extern "C" void kernel_launch(void* const* d_in, const int* in_sizes, int n_in,
                              void* d_out, int out_size, void* d_ws, size_t ws_size,
                              hipStream_t stream) {
  (void)in_sizes; (void)n_in; (void)out_size; (void)ws_size;

  const float* q    = (const float*)d_in[0];
  const float* k    = (const float*)d_in[1];
  const float* v    = (const float*)d_in[2];
  const float* Wq_w = (const float*)d_in[3];
  const float* Wq_b = (const float*)d_in[4];
  const float* Wk_w = (const float*)d_in[5];
  const float* Wk_b = (const float*)d_in[6];
  const float* Wv_w = (const float*)d_in[7];
  const float* Wv_b = (const float*)d_in[8];
  const float* fc_w = (const float*)d_in[9];
  const float* fc_b = (const float*)d_in[10];
  const float* E    = (const float*)d_in[11];

  char* ws = (char*)d_ws;
  __bf16* Qbf = (__bf16*)(ws + 0);                      // 4 MiB
  __bf16* Kbf = (__bf16*)(ws + 4u * 1024 * 1024);       // 4 MiB
  __bf16* Vbf = (__bf16*)(ws + 8u * 1024 * 1024);       // 4 MiB
  __bf16* Ebf = (__bf16*)(ws + 12u * 1024 * 1024);      // 256 KiB
  float*  Obuf = (float*)(ws + 12u * 1024 * 1024 + 512u * 1024);  // 8 MiB

  float* out  = (float*)d_out;
  float* attn = out + (size_t)NBATCH * SEQL * DMODEL;

  k_f32_to_bf16<<<(SEQL * DHEAD) / 256, 256, 0, stream>>>(E, Ebf, SEQL * DHEAD);

  dim3 gp(32, 8);
  k_gemm512<true><<<gp, 256, 0, stream>>>(q, Wq_w, Wq_b, (void*)Qbf);
  k_gemm512<true><<<gp, 256, 0, stream>>>(k, Wk_w, Wk_b, (void*)Kbf);
  k_gemm512<true><<<gp, 256, 0, stream>>>(v, Wv_w, Wv_b, (void*)Vbf);

  k_logits_fused<<<dim3(128, 16, 16), 256, 0, stream>>>(Qbf, Kbf, Ebf, attn);

  k_softmax<<<NBH * SEQL, 256, 0, stream>>>(attn);

  k_av<<<dim3(64, 16), 256, 0, stream>>>(attn, Vbf, Obuf);

  k_gemm512<false><<<gp, 256, 0, stream>>>(Obuf, fc_w, fc_b, (void*)out);
}
